// CRN_21852793602684
// MI455X (gfx1250) — matrix operand order repaired and checked by reading, not compiled, not run on hardware
//
#include <hip/hip_runtime.h>
#include <hip/hip_bf16.h>
#include <math.h>

typedef __attribute__((ext_vector_type(16))) _Float16 v16h;
typedef __attribute__((ext_vector_type(8)))  float    v8f;

#define B_   1024
#define S_   512
#define DX   128
#define DH   128
#define G3   384      // 3*DH
#define KT   4        // 128/32 K tiles
#define NTR  24       // 384/16 N tiles (recurrent: W_hh only)
#define NTH  16       // 256/16 N tiles (heads: tW1 | yW1_h)
#define GPR  388      // k_rec gout pitch (384 + pad)
#define HGP  260      // k_hd gout pitch (256 + pad)

__device__ __forceinline__ float gelu_exact(float v) {
    return 0.5f * v * (1.0f + erff(v * 0.70710678118654752f));
}
__device__ __forceinline__ float sigm(float v) {
    return 1.0f / (1.0f + __expf(-v));
}

// ---------------- one-time precompute kernels ----------------

// x_enc = x @ xW.T + xb       grid(B), block(128)
__global__ void k_xenc(const float* __restrict__ x, const float* __restrict__ xW,
                       const float* __restrict__ xb, float* __restrict__ xe) {
    int b = blockIdx.x, k = threadIdx.x;
    const float* xr = x + b * DX;
    const float* wr = xW + k * DX;
    float s = 0.f;
    for (int i = 0; i < DX; ++i) s += xr[i] * wr[i];
    xe[b * DX + k] = s + xb[k];
}

// h0 = tanh(x_enc @ h0W.T + h0b)   grid(B), block(128)
__global__ void k_h0(const float* __restrict__ xe, const float* __restrict__ h0W,
                     const float* __restrict__ h0b, float* __restrict__ h0g) {
    int b = blockIdx.x, k = threadIdx.x;
    const float* xr = xe + b * DX;
    const float* wr = h0W + k * DX;
    float s = 0.f;
    for (int i = 0; i < DX; ++i) s += xr[i] * wr[i];
    h0g[b * DH + k] = tanhf(s + h0b[k]);
}

// gxx = x_enc @ W_ih[:, :128].T + b_ih     grid(B), block(384)
__global__ void k_gxx(const float* __restrict__ xe, const float* __restrict__ W_ih,
                      const float* __restrict__ b_ih, float* __restrict__ gxx) {
    int b = blockIdx.x, n = threadIdx.x;
    const float* xr = xe + b * DX;
    const float* wr = W_ih + n * 177;
    float s = 0.f;
    for (int i = 0; i < DX; ++i) s += xr[i] * wr[i];
    gxx[b * G3 + n] = s + b_ih[n];
}

// yx = x_enc @ yW1[:, 128:256].T + yb1     grid(B), block(128)
__global__ void k_yx(const float* __restrict__ xe, const float* __restrict__ yW1,
                     const float* __restrict__ yb1, float* __restrict__ yx) {
    int b = blockIdx.x, k = threadIdx.x;
    const float* xr = xe + b * DX;
    const float* wr = yW1 + k * 304 + 128;
    float s = 0.f;
    for (int i = 0; i < DX; ++i) s += xr[i] * wr[i];
    yx[b * DH + k] = s + yb1[k];
}

// tabAg[v][n] = a_emb[v] . W_ih[n][128:160]    grid(100), block(384)
__global__ void k_tabAg(const float* __restrict__ a_emb, const float* __restrict__ W_ih,
                        float* __restrict__ tabAg) {
    int v = blockIdx.x, n = threadIdx.x;
    const float* er = a_emb + v * 32;
    const float* wr = W_ih + n * 177 + 128;
    float s = 0.f;
    for (int i = 0; i < 32; ++i) s += er[i] * wr[i];
    tabAg[v * G3 + n] = s;
}

// tabTg[v][n] = t_emb[v] . W_ih[n][160:176]    grid(4), block(384)
__global__ void k_tabTg(const float* __restrict__ t_emb, const float* __restrict__ W_ih,
                        float* __restrict__ tabTg) {
    int v = blockIdx.x, n = threadIdx.x;
    const float* er = t_emb + v * 16;
    const float* wr = W_ih + n * 177 + 160;
    float s = 0.f;
    for (int i = 0; i < 16; ++i) s += er[i] * wr[i];
    tabTg[v * G3 + n] = s;
}

// tabAy[v][k] = a_emb[v] . yW1[k][256:288]     grid(100), block(128)
__global__ void k_tabAy(const float* __restrict__ a_emb, const float* __restrict__ yW1,
                        float* __restrict__ tabAy) {
    int v = blockIdx.x, k = threadIdx.x;
    const float* er = a_emb + v * 32;
    const float* wr = yW1 + k * 304 + 256;
    float s = 0.f;
    for (int i = 0; i < 32; ++i) s += er[i] * wr[i];
    tabAy[v * DH + k] = s;
}

// tabTy[v][k] = t_emb[v] . yW1[k][288:304]     grid(4), block(128)
__global__ void k_tabTy(const float* __restrict__ t_emb, const float* __restrict__ yW1,
                        float* __restrict__ tabTy) {
    int v = blockIdx.x, k = threadIdx.x;
    const float* er = t_emb + v * 16;
    const float* wr = yW1 + k * 304 + 288;
    float s = 0.f;
    for (int i = 0; i < 16; ++i) s += er[i] * wr[i];
    tabTy[v * DH + k] = s;
}

// wyg[n] = W_ih[n][176]      grid(1), block(384)
__global__ void k_wyg(const float* __restrict__ W_ih, float* __restrict__ wyg) {
    int n = threadIdx.x;
    wyg[n] = W_ih[n * 177 + 176];
}

// Pack W_hh.T [k=0..127][n=0..383] into WMMA B-fragment order (f16):
//   idx = (((kt*NTR + nt)*32 + lane)*16 + e), n = 16*nt + (lane&15),
//   K = 32*kt + 8*(lane>>4) + e + 8*(e>>3).
// grid(KT*NTR=96), block(512): tid = lane*16 + e.
__global__ void k_wcat_hh(const float* __restrict__ W_hh, _Float16* __restrict__ wcat) {
    int kt = blockIdx.x / NTR;
    int nt = blockIdx.x % NTR;
    int lane = threadIdx.x >> 4;
    int e    = threadIdx.x & 15;
    int k = 32 * kt + 8 * (lane >> 4) + e + 8 * (e >> 3);
    int n = 16 * nt + (lane & 15);
    float v = W_hh[n * DX + k];                            // gh = h @ W_hh.T
    wcat[(((size_t)(kt * NTR + nt)) * 32 + lane) * 16 + e] = (_Float16)v;
}

// Pack [tW1.T | yW1_h.T] [k=0..127][n=0..255] in fragment order. grid(KT*NTH=64), block(512).
__global__ void k_wcat_hd(const float* __restrict__ tW1, const float* __restrict__ yW1,
                          _Float16* __restrict__ wcat) {
    int kt = blockIdx.x / NTH;
    int nt = blockIdx.x % NTH;
    int lane = threadIdx.x >> 4;
    int e    = threadIdx.x & 15;
    int k = 32 * kt + 8 * (lane >> 4) + e + 8 * (e >> 3);
    int n = 16 * nt + (lane & 15);
    float v;
    if (n < 128) v = tW1[n * DX + k];                      // th = h @ tW1.T
    else         v = yW1[(n - 128) * 304 + k];             // yy_h = h @ yW1[:, :128].T
    wcat[(((size_t)(kt * NTH + nt)) * 32 + lane) * 16 + e] = (_Float16)v;
}

// ---------------- recurrent kernel (GRU only) ----------------
// grid(64) x block(256 = 8 waves): WG owns 16 batch rows, all 512 steps.
// Each wave owns 3 N-tiles -> 12 resident B fragments (96 VGPRs).
__global__ __launch_bounds__(256) void k_rec(
    const int* __restrict__ a, const int* __restrict__ t,
    const float* __restrict__ y, const float* __restrict__ mask,
    const float* __restrict__ b_hh,
    const float* __restrict__ gxx, const float* __restrict__ tabAg,
    const float* __restrict__ tabTg, const float* __restrict__ wyg,
    const _Float16* __restrict__ wcat, const float* __restrict__ h0g,
    float* __restrict__ h_out)
{
    __shared__ float gout[16 * GPR];                    // gh, f32
    __shared__ float hcur[16 * DH];                     // current h, f32
    __shared__ __align__(32) _Float16 hA[KT * 32 * 16]; // h as pre-swizzled A fragments
    __shared__ float s_bhh[G3];
    __shared__ int   sc_a[16], sc_t[16];
    __shared__ float sc_y[16], sc_m[16];

    const int tid   = threadIdx.x;
    const int lane  = tid & 31;
    const int w     = tid >> 5;             // wave id 0..7
    const int bbase = blockIdx.x * 16;

    for (int i = tid; i < G3; i += 256) s_bhh[i] = b_hh[i];

    // init h from h0, write both layouts
#pragma unroll
    for (int i = 0; i < 8; ++i) {
        int idx = tid + 256 * i;
        int m = idx >> 7, j = idx & 127;
        float h = h0g[(bbase + m) * DH + j];
        hcur[m * DH + j] = h;
        int kt = j >> 5, k = j & 31;
        int hi = (k >> 3) & 1;
        int rem = k - 8 * hi;
        int vv = ((rem >> 4) & 1) * 4 + ((rem >> 1) & 3);
        int e = 2 * vv + (rem & 1);
        hA[(kt * 32 + (hi * 16 + m)) * 16 + e] = (_Float16)h;
    }

    // W_hh B fragments: step-invariant, resident (12 x v16h = 96 VGPRs).
    v16h Bf[12];
#pragma unroll
    for (int q = 0; q < 3; ++q) {
        int nt = w * 3 + q;
#pragma unroll
        for (int kt = 0; kt < 4; ++kt)
            Bf[q * 4 + kt] = *(const v16h*)(wcat + (((size_t)(kt * NTR + nt)) * 32 + lane) * 16);
    }
    __syncthreads();

    for (int s = 0; s < S_; ++s) {
        // stage per-step scalars for the 16 rows
        if (tid < 16) {
            size_t off = (size_t)(bbase + tid) * S_ + s;
            sc_a[tid] = a[off];
            sc_t[tid] = t[off];
            sc_y[tid] = y[off];
            sc_m[tid] = mask[off];
        }
        // gh = h @ W_hh.T : [16,128] x [128,384] via 12 WMMAs per wave
        v16h A[4];
#pragma unroll
        for (int kt = 0; kt < 4; ++kt)
            A[kt] = *(const v16h*)(hA + (kt * 32 + lane) * 16);
#pragma unroll
        for (int q = 0; q < 3; ++q) {
            v8f acc = {};
#pragma unroll
            for (int kt = 0; kt < 4; ++kt)
                acc = __builtin_amdgcn_wmma_f32_16x16x32_f16(
                    false, A[kt], false, Bf[q * 4 + kt], (short)0, acc, false, false);
            int nt = w * 3 + q;
            int n0 = nt * 16 + (lane & 15);
            int mb = (lane >> 4) * 8;
#pragma unroll
            for (int v = 0; v < 8; ++v)
                gout[(mb + v) * GPR + n0] = acc[v];
        }
        __syncthreads();

        // GRU elementwise update (8 elems/thread)
#pragma unroll
        for (int i = 0; i < 8; ++i) {
            int idx = tid + 256 * i;
            int m = idx >> 7, j = idx & 127;
            int b = bbase + m;
            int av = sc_a[m], tv = sc_t[m];
            float yv = sc_y[m], mv = sc_m[m];
            float hold = hcur[m * DH + j];
            h_out[((size_t)b * S_ + s) * DH + j] = hold;  // h_seq = pre-update h

            float gxr = gxx[b * G3 + j]       + tabAg[av * G3 + j]       + tabTg[tv * G3 + j]       + yv * wyg[j];
            float gxz = gxx[b * G3 + 128 + j] + tabAg[av * G3 + 128 + j] + tabTg[tv * G3 + 128 + j] + yv * wyg[128 + j];
            float gxn = gxx[b * G3 + 256 + j] + tabAg[av * G3 + 256 + j] + tabTg[tv * G3 + 256 + j] + yv * wyg[256 + j];
            float ghr = gout[m * GPR + j]       + s_bhh[j];
            float ghz = gout[m * GPR + 128 + j] + s_bhh[128 + j];
            float ghn = gout[m * GPR + 256 + j] + s_bhh[256 + j];
            float r = sigm(gxr + ghr);
            float z = sigm(gxz + ghz);
            float n = tanhf(gxn + r * ghn);
            float hnext = (1.f - z) * n + z * hold;
            float hnew = mv * hnext + (1.f - mv) * hold;
            hcur[m * DH + j] = hnew;
            int kt = j >> 5, k = j & 31;
            int hi = (k >> 3) & 1;
            int rem = k - 8 * hi;
            int vv = ((rem >> 4) & 1) * 4 + ((rem >> 1) & 3);
            int e = 2 * vv + (rem & 1);
            hA[(kt * 32 + (hi * 16 + m)) * 16 + e] = (_Float16)hnew;
        }
        __syncthreads();
    }
}

// ---------------- head kernel (fully parallel over B*S) ----------------
// grid(B*S/16 = 32768) x block(256 = 8 waves). Each WG: 16 rows of flattened
// (b,s), GEMM [16,128]x[128,256] + gelu + shuffle-reduced layer-2.
__global__ __launch_bounds__(256) void k_hd(
    const int* __restrict__ a, const int* __restrict__ t,
    const float* __restrict__ tb1, const float* __restrict__ tW2,
    const float* __restrict__ tb2, const float* __restrict__ yW2,
    const float* __restrict__ yb2, const float* __restrict__ yx,
    const float* __restrict__ tabAy, const float* __restrict__ tabTy,
    const _Float16* __restrict__ wcat, const float* __restrict__ h_seq,
    float* __restrict__ y_out, float* __restrict__ t_out)
{
    __shared__ float gout[16 * HGP];                    // th | yy, f32
    __shared__ __align__(32) _Float16 hA[KT * 32 * 16];

    const int tid  = threadIdx.x;
    const int lane = tid & 31;
    const int w    = tid >> 5;
    const size_t r0 = (size_t)blockIdx.x * 16;          // flattened (b*S + s) base
    const int b = (int)(r0 >> 9);                       // S_ = 512; tile stays in one b

    // load 16 h rows, build A fragments
#pragma unroll
    for (int i = 0; i < 8; ++i) {
        int idx = tid + 256 * i;
        int m = idx >> 7, j = idx & 127;
        float h = h_seq[(r0 + m) * DH + j];
        int kt = j >> 5, k = j & 31;
        int hi = (k >> 3) & 1;
        int rem = k - 8 * hi;
        int vv = ((rem >> 4) & 1) * 4 + ((rem >> 1) & 3);
        int e = 2 * vv + (rem & 1);
        hA[(kt * 32 + (hi * 16 + m)) * 16 + e] = (_Float16)h;
    }
    __syncthreads();

    v16h A[4];
#pragma unroll
    for (int kt = 0; kt < 4; ++kt)
        A[kt] = *(const v16h*)(hA + (kt * 32 + lane) * 16);
#pragma unroll
    for (int q = 0; q < 2; ++q) {
        int nt = w * 2 + q;
        v8f acc = {};
#pragma unroll
        for (int kt = 0; kt < 4; ++kt) {
            v16h Bfr = *(const v16h*)(wcat + (((size_t)(kt * NTH + nt)) * 32 + lane) * 16);
            acc = __builtin_amdgcn_wmma_f32_16x16x32_f16(
                false, A[kt], false, Bfr, (short)0, acc, false, false);
        }
        int n0 = nt * 16 + (lane & 15);
        int mb = (lane >> 4) * 8;
#pragma unroll
        for (int v = 0; v < 8; ++v)
            gout[(mb + v) * HGP + n0] = acc[v];
    }
    __syncthreads();

    // gelu on both head hiddens
#pragma unroll
    for (int i = 0; i < 8; ++i) {
        int idx = tid + 256 * i;
        int m = idx >> 7, j = idx & 127;
        size_t r = r0 + m;
        int av = a[r], tv = t[r];
        float th = gout[m * HGP + j] + tb1[j];
        gout[m * HGP + j] = gelu_exact(th);
        float yy = gout[m * HGP + 128 + j] + yx[b * DH + j]
                 + tabAy[av * DH + j] + tabTy[tv * DH + j];
        gout[m * HGP + 128 + j] = gelu_exact(yy);
    }
    __syncthreads();

    // layer-2: wave w handles rows 2w and 2w+1, shuffle butterfly reduce
#pragma unroll
    for (int rr = 0; rr < 2; ++rr) {
        int m = w * 2 + rr;
        float p0 = 0.f, p1 = 0.f, p2 = 0.f, p3 = 0.f, py = 0.f;
#pragma unroll
        for (int c = 0; c < 4; ++c) {
            int k2 = lane + 32 * c;
            float g  = gout[m * HGP + k2];
            float gy = gout[m * HGP + 128 + k2];
            p0 += g * tW2[0 * 128 + k2];
            p1 += g * tW2[1 * 128 + k2];
            p2 += g * tW2[2 * 128 + k2];
            p3 += g * tW2[3 * 128 + k2];
            py += gy * yW2[k2];
        }
#pragma unroll
        for (int off = 16; off > 0; off >>= 1) {
            p0 += __shfl_xor(p0, off, 32);
            p1 += __shfl_xor(p1, off, 32);
            p2 += __shfl_xor(p2, off, 32);
            p3 += __shfl_xor(p3, off, 32);
            py += __shfl_xor(py, off, 32);
        }
        if (lane == 0) {
            size_t r = r0 + m;
            t_out[r * 4 + 0] = p0 + tb2[0];
            t_out[r * 4 + 1] = p1 + tb2[1];
            t_out[r * 4 + 2] = p2 + tb2[2];
            t_out[r * 4 + 3] = p3 + tb2[3];
            y_out[r]         = py + yb2[0];
        }
    }
}

// ---------------- host launcher ----------------
extern "C" void kernel_launch(void* const* d_in, const int* in_sizes, int n_in,
                              void* d_out, int out_size, void* d_ws, size_t ws_size,
                              hipStream_t stream) {
    const float* x     = (const float*)d_in[0];
    const int*   a     = (const int*)  d_in[1];
    const int*   t     = (const int*)  d_in[2];
    const float* y     = (const float*)d_in[3];
    const float* mask  = (const float*)d_in[4];
    const float* xW    = (const float*)d_in[5];
    const float* xb    = (const float*)d_in[6];
    const float* a_emb = (const float*)d_in[7];
    const float* t_emb = (const float*)d_in[8];
    const float* W_ih  = (const float*)d_in[9];
    const float* b_ih  = (const float*)d_in[10];
    const float* W_hh  = (const float*)d_in[11];
    const float* b_hh  = (const float*)d_in[12];
    const float* h0W   = (const float*)d_in[13];
    const float* h0b   = (const float*)d_in[14];
    const float* tW1   = (const float*)d_in[15];
    const float* tb1   = (const float*)d_in[16];
    const float* tW2   = (const float*)d_in[17];
    const float* tb2   = (const float*)d_in[18];
    const float* yW1   = (const float*)d_in[19];
    const float* yb1   = (const float*)d_in[20];
    const float* yW2   = (const float*)d_in[21];
    const float* yb2   = (const float*)d_in[22];

    // workspace carve-up (256B aligned)
    char* p = (char*)d_ws;
    auto carve = [&](size_t bytes) -> void* {
        void* r = (void*)p;
        p += (bytes + 255) & ~(size_t)255;
        return r;
    };
    float*    xe      = (float*)   carve((size_t)B_ * DX * 4);
    float*    h0g     = (float*)   carve((size_t)B_ * DH * 4);
    float*    gxx     = (float*)   carve((size_t)B_ * G3 * 4);
    float*    yxp     = (float*)   carve((size_t)B_ * DH * 4);
    float*    tabAg   = (float*)   carve((size_t)100 * G3 * 4);
    float*    tabTg   = (float*)   carve((size_t)4 * G3 * 4);
    float*    tabAy   = (float*)   carve((size_t)100 * DH * 4);
    float*    tabTy   = (float*)   carve((size_t)4 * DH * 4);
    float*    wyg     = (float*)   carve((size_t)G3 * 4);
    _Float16* wcat_hh = (_Float16*)carve((size_t)KT * NTR * 32 * 16 * 2);
    _Float16* wcat_hd = (_Float16*)carve((size_t)KT * NTH * 32 * 16 * 2);

    float* y_out = (float*)d_out;                          // [B,S,1]
    float* t_out = y_out + (size_t)B_ * S_;                // [B,S,4]
    float* h_out = y_out + (size_t)B_ * S_ * 5;            // [B,S,128]

    k_xenc   <<<B_, 128, 0, stream>>>(x, xW, xb, xe);
    k_h0     <<<B_, 128, 0, stream>>>(xe, h0W, h0b, h0g);
    k_gxx    <<<B_, G3,  0, stream>>>(xe, W_ih, b_ih, gxx);
    k_yx     <<<B_, 128, 0, stream>>>(xe, yW1, yb1, yxp);
    k_tabAg  <<<100, G3, 0, stream>>>(a_emb, W_ih, tabAg);
    k_tabTg  <<<4,   G3, 0, stream>>>(t_emb, W_ih, tabTg);
    k_tabAy  <<<100, 128, 0, stream>>>(a_emb, yW1, tabAy);
    k_tabTy  <<<4,   128, 0, stream>>>(t_emb, yW1, tabTy);
    k_wyg    <<<1,   G3,  0, stream>>>(W_ih, wyg);
    k_wcat_hh<<<KT * NTR, 512, 0, stream>>>(W_hh, wcat_hh);
    k_wcat_hd<<<KT * NTH, 512, 0, stream>>>(tW1, yW1, wcat_hd);

    // sequential GRU (writes h_seq)
    k_rec<<<B_ / 16, 256, 0, stream>>>(
        a, t, y, mask, b_hh, gxx, tabAg, tabTg, wyg, wcat_hh, h0g, h_out);

    // fully parallel heads over all B*S rows (reads h_seq)
    k_hd<<<(B_ * S_) / 16, 256, 0, stream>>>(
        a, t, tb1, tW2, tb2, yW2, yb2, yxp, tabAy, tabTy, wcat_hd, h_out,
        y_out, t_out);
}
